// HeteroGraphAE_66340064854258
// MI455X (gfx1250) — matrix operand order repaired and checked by reading, not compiled, not run on hardware
//
#include <hip/hip_runtime.h>
#include <hip/hip_bf16.h>

typedef __bf16 bf16_t;
typedef __attribute__((ext_vector_type(16))) __bf16 v16bf;
typedef __attribute__((ext_vector_type(8)))  __bf16 v8bf;
typedef __attribute__((ext_vector_type(8)))  float   v8f;

#define NN    20000
#define HH    512
#define LL    256
#define HEADS 8
#define DHD   64

// ---------------------------------------------------------------- utilities

__global__ void fill_f32_kernel(float* __restrict__ p, float v, int n) {
  int i = blockIdx.x * blockDim.x + threadIdx.x;
  if (i < n) p[i] = v;
}

__global__ void f32_to_bf16_kernel(const float* __restrict__ in, bf16_t* __restrict__ out, int n) {
  int i = blockIdx.x * blockDim.x + threadIdx.x;
  if (i < n) out[i] = (bf16_t)in[i];
}

// W [K x Nout] row-major f32  ->  Wt [Nout x K] bf16 (WMMA-ready B operand)
__global__ void convert_w_transpose_kernel(const float* __restrict__ W, bf16_t* __restrict__ Wt,
                                           int K, int Nout) {
  int i = blockIdx.x * blockDim.x + threadIdx.x;
  if (i >= K * Nout) return;
  int k = i / Nout, nc = i % Nout;                 // coalesced read
  Wt[(size_t)nc * K + k] = (bf16_t)W[i];
}

// ---------------------------------------------------------------- WMMA GEMM
// C[M x Nout] = A[M x K](bf16 rm) @ Bt[Nout x K](bf16, pre-transposed) (+bias)
// one wave computes a 32x32 tile via 2x2 v_wmma_f32_16x16x32_bf16
__global__ __launch_bounds__(256) void gemm_bf16_wmma_kernel(
    const bf16_t* __restrict__ A, const bf16_t* __restrict__ Bt,
    const float* __restrict__ bias, float* __restrict__ C,
    int M, int K, int Nout) {
  const int lane = threadIdx.x & 31;
  const int wid  = threadIdx.x >> 5;
  const int tilesN = Nout >> 5;
  const int tilesM = M >> 5;
  int tile = blockIdx.x * 8 + wid;
  if (tile >= tilesM * tilesN) return;
  const int tm = tile / tilesN;
  const int tn = tile % tilesN;

  const int half = lane >> 4;          // K-half selector (A/B operand layout)
  const int l16  = lane & 15;

  const bf16_t* Arow0 = A + (size_t)(tm * 32 + l16) * K;       // M = lane&15
  const bf16_t* Arow1 = Arow0 + (size_t)16 * K;
  const bf16_t* Bc0   = Bt + (size_t)(tn * 32 + l16) * K;      // N = lane&15
  const bf16_t* Bc1   = Bc0 + (size_t)16 * K;

  v8f acc00 = {0.f,0.f,0.f,0.f,0.f,0.f,0.f,0.f};
  v8f acc01 = acc00, acc10 = acc00, acc11 = acc00;

  for (int k0 = 0; k0 < K; k0 += 32) {
    const int ka = k0 + half * 16;     // A: elems 0-15 = K[ka..ka+15] sequential
    const int kb = k0 + half * 16;     // B: elems 0-15 = K[kb..kb+15] sequential
    v16bf a0 = *(const v16bf*)(Arow0 + ka);
    v16bf a1 = *(const v16bf*)(Arow1 + ka);
    v16bf b0 = *(const v16bf*)(Bc0 + kb);
    v16bf b1 = *(const v16bf*)(Bc1 + kb);
    acc00 = __builtin_amdgcn_wmma_f32_16x16x32_bf16(false, a0, false, b0, (short)0, acc00, false, false);
    acc01 = __builtin_amdgcn_wmma_f32_16x16x32_bf16(false, a0, false, b1, (short)0, acc01, false, false);
    acc10 = __builtin_amdgcn_wmma_f32_16x16x32_bf16(false, a1, false, b0, (short)0, acc10, false, false);
    acc11 = __builtin_amdgcn_wmma_f32_16x16x32_bf16(false, a1, false, b1, (short)0, acc11, false, false);
  }

  const int col0 = tn * 32 + l16;
  const int col1 = col0 + 16;
  const float bias0 = bias ? bias[col0] : 0.f;
  const float bias1 = bias ? bias[col1] : 0.f;
  const int mrow = tm * 32 + half * 8;           // C: row = m0 + 8*half + r
#pragma unroll
  for (int r = 0; r < 8; ++r) {
    C[(size_t)(mrow + r)      * Nout + col0] = acc00[r] + bias0;
    C[(size_t)(mrow + r)      * Nout + col1] = acc01[r] + bias1;
    C[(size_t)(mrow + 16 + r) * Nout + col0] = acc10[r] + bias0;
    C[(size_t)(mrow + 16 + r) * Nout + col1] = acc11[r] + bias1;
  }
}

// ---------------------------------------------------------------- GCN pieces

__global__ void deg_accum_kernel(const int* __restrict__ dst, float* __restrict__ deg, int E) {
  int i = blockIdx.x * blockDim.x + threadIdx.x;
  if (i < E) atomicAdd(&deg[dst[i]], 1.0f);
}

__global__ void finalize_dinv_kernel(float* __restrict__ deg, int n) {
  int i = blockIdx.x * blockDim.x + threadIdx.x;
  if (i < n) { float d = deg[i]; deg[i] = d > 0.f ? rsqrtf(d) : 0.f; }
}

// out[d] += h[s] * dinv[s]*dinv[d], edges + implicit self loops; one wave/edge
__global__ __launch_bounds__(256) void gcn_scatter_kernel(
    const float* __restrict__ H, const float* __restrict__ dinv,
    const int* __restrict__ src, const int* __restrict__ dst,
    float* __restrict__ out, int E, int n, int C) {
  int lane = threadIdx.x & 31;
  int gw = (blockIdx.x * blockDim.x + threadIdx.x) >> 5;
  int nw = (gridDim.x * blockDim.x) >> 5;
  int total = E + n;
  for (int e = gw; e < total; e += nw) {
    int s, d;
    if (e < E) { s = src[e]; d = dst[e]; } else { s = e - E; d = s; }
    float w = dinv[s] * dinv[d];
    const float4* hs = (const float4*)(H + (size_t)s * C);
    float* od = out + (size_t)d * C;
    for (int j = lane; j < (C >> 2); j += 32) {
      float4 v = hs[j];
      atomicAdd(&od[j*4+0], v.x * w);
      atomicAdd(&od[j*4+1], v.y * w);
      atomicAdd(&od[j*4+2], v.z * w);
      atomicAdd(&od[j*4+3], v.w * w);
    }
  }
}

// ---------------------------------------------------------------- BN + SiLU

__global__ __launch_bounds__(512) void bn_reduce_kernel(const float* __restrict__ X,
                                                        float* __restrict__ stats, int n) {
  int c = threadIdx.x;                       // blockDim.x == 512 == C
  int r0 = blockIdx.x * 50;
  int r1 = r0 + 50; if (r1 > n) r1 = n;
  float s = 0.f, s2 = 0.f;
  for (int r = r0; r < r1; ++r) { float v = X[(size_t)r * HH + c]; s += v; s2 += v * v; }
  atomicAdd(&stats[c], s);
  atomicAdd(&stats[HH + c], s2);
}

__global__ void bn_silu_bf16_kernel(const float* __restrict__ X, const float* __restrict__ stats,
                                    const float* __restrict__ gamma, const float* __restrict__ beta,
                                    bf16_t* __restrict__ Y, int n) {
  int i = blockIdx.x * blockDim.x + threadIdx.x;
  if (i >= n * HH) return;
  int c = i & (HH - 1);
  float inv_n = 1.0f / (float)n;
  float mu = stats[c] * inv_n;
  float var = stats[HH + c] * inv_n - mu * mu;       // biased var (jnp.var)
  float sc = gamma[c] * rsqrtf(var + 1e-5f);
  float sh = beta[c] - mu * sc;
  float v = X[i] * sc + sh;
  float y = v / (1.f + __expf(-v));                   // SiLU
  Y[i] = (bf16_t)y;
}

// ---------------------------------------------------------------- GATv2

__device__ inline unsigned encF(float f) {
  unsigned u = __float_as_uint(f);
  return (u & 0x80000000u) ? ~u : (u | 0x80000000u);
}
__device__ inline float decF(unsigned u) {
  u = (u & 0x80000000u) ? (u ^ 0x80000000u) : ~u;
  return __uint_as_float(u);
}

// per edge: logits[e,h] = sum_k att[h,k]*lrelu(hl[s,h,k]+hr[d,h,k]); atomic max per (d,h)
__global__ __launch_bounds__(256) void gat_logits_kernel(
    const float* __restrict__ HL, const float* __restrict__ HR,
    const float* __restrict__ att, const int* __restrict__ src, const int* __restrict__ dst,
    int E, int n, float* __restrict__ logits, unsigned* __restrict__ mEnc) {
  int lane = threadIdx.x & 31;
  int gw = (blockIdx.x * blockDim.x + threadIdx.x) >> 5;
  int nw = (gridDim.x * blockDim.x) >> 5;
  int total = E + n;
  int head = lane >> 2, part = lane & 3;              // 4 lanes per head
  for (int e = gw; e < total; e += nw) {
    int s, d;
    if (e < E) { s = src[e]; d = dst[e]; } else { s = e - E; d = s; }
    const float* hl = HL + (size_t)s * HH + head * DHD + part * 16;
    const float* hr = HR + (size_t)d * HH + head * DHD + part * 16;
    const float* at = att + head * DHD + part * 16;
    float acc = 0.f;
#pragma unroll
    for (int j = 0; j < 16; ++j) {
      float v = hl[j] + hr[j];
      v = v > 0.f ? v : 0.2f * v;                     // leaky_relu(0.2)
      acc += at[j] * v;
    }
    acc += __shfl_xor(acc, 1, 32);
    acc += __shfl_xor(acc, 2, 32);
    if (part == 0) {
      logits[(size_t)e * HEADS + head] = acc;
      atomicMax(&mEnc[(size_t)d * HEADS + head], encF(acc));
    }
  }
}

// a = exp(logit - m[d]); denom[d,h] += a ; logits overwritten with a
__global__ void gat_exp_denom_kernel(float* __restrict__ logits, const unsigned* __restrict__ mEnc,
                                     float* __restrict__ denom, const int* __restrict__ dst,
                                     int E, int n) {
  int i = blockIdx.x * blockDim.x + threadIdx.x;
  if (i >= (E + n) * HEADS) return;
  int e = i >> 3, h = i & 7;
  int d = (e < E) ? dst[e] : (e - E);
  float a = __expf(logits[i] - decF(mEnc[(size_t)d * HEADS + h]));
  logits[i] = a;
  atomicAdd(&denom[(size_t)d * HEADS + h], a);
}

// out[d,:] += (a[e,h]/denom[d,h]) * hl[s,h,:] ; one wave per edge
__global__ __launch_bounds__(256) void gat_scatter_kernel(
    const float* __restrict__ HL, const float* __restrict__ aV, const float* __restrict__ denom,
    const int* __restrict__ src, const int* __restrict__ dst,
    int E, int n, float* __restrict__ out) {
  int lane = threadIdx.x & 31;
  int gw = (blockIdx.x * blockDim.x + threadIdx.x) >> 5;
  int nw = (gridDim.x * blockDim.x) >> 5;
  int total = E + n;
  for (int e = gw; e < total; e += nw) {
    int s, d;
    if (e < E) { s = src[e]; d = dst[e]; } else { s = e - E; d = s; }
#pragma unroll
    for (int j = 0; j < 4; ++j) {
      int c0 = j * 128 + lane * 4;
      int h = c0 >> 6;
      float w = aV[(size_t)e * HEADS + h] / denom[(size_t)d * HEADS + h];
      float4 v = *(const float4*)(HL + (size_t)s * HH + c0);
      float* od = out + (size_t)d * HH + c0;
      atomicAdd(&od[0], v.x * w);
      atomicAdd(&od[1], v.y * w);
      atomicAdd(&od[2], v.z * w);
      atomicAdd(&od[3], v.w * w);
    }
  }
}

__global__ void init_out_bias_kernel(float* __restrict__ out, const float* __restrict__ ba,
                                     const float* __restrict__ br, int total) {
  int i = blockIdx.x * blockDim.x + threadIdx.x;
  if (i < total) { int c = i & (LL - 1); out[i] = ba[c] + br[c]; }
}

// ---------------------------------------------------------------- launch

static inline unsigned cdiv(long long a, long long b) { return (unsigned)((a + b - 1) / b); }

extern "C" void kernel_launch(void* const* d_in, const int* in_sizes, int n_in,
                              void* d_out, int out_size, void* d_ws, size_t ws_size,
                              hipStream_t stream) {
  const float* x       = (const float*)d_in[0];
  const int*   src_a   = (const int*)d_in[1];
  const int*   dst_a   = src_a + (in_sizes[1] / 2);
  const int*   src_r   = (const int*)d_in[2];
  const int*   dst_r   = src_r + (in_sizes[2] / 2);
  const float* W1a = (const float*)d_in[3];
  const float* W1r = (const float*)d_in[5];
  const float* g1  = (const float*)d_in[7];
  const float* be1 = (const float*)d_in[8];
  const float* Wla = (const float*)d_in[9];
  const float* bla = (const float*)d_in[10];
  const float* Wra = (const float*)d_in[11];
  const float* bra = (const float*)d_in[12];
  const float* atta= (const float*)d_in[13];
  const float* Wlr = (const float*)d_in[15];
  const float* blr = (const float*)d_in[16];
  const float* Wrr = (const float*)d_in[17];
  const float* brr = (const float*)d_in[18];
  const float* attr= (const float*)d_in[19];
  const float* g2  = (const float*)d_in[21];
  const float* be2 = (const float*)d_in[22];
  const float* Wza = (const float*)d_in[23];
  const float* bza = (const float*)d_in[24];
  const float* Wzr = (const float*)d_in[25];
  const float* bzr = (const float*)d_in[26];

  const int N = NN, E = in_sizes[1] / 2, H = HH, L = LL;
  const int TOT = E + N;

  // workspace carve-up
  char* base = (char*)d_ws;
  size_t off = 0;
  auto carve = [&](size_t bytes) { char* p = base + off; off = (off + bytes + 255) & ~(size_t)255; return p; };
  float*  dinv_a = (float*)carve((size_t)N * 4);
  float*  dinv_r = (float*)carve((size_t)N * 4);
  float*  stats  = (float*)carve((size_t)2 * H * 4);
  unsigned* mEnc = (unsigned*)carve((size_t)N * HEADS * 4);
  float*  denom  = (float*)carve((size_t)N * HEADS * 4);
  float*  logits = (float*)carve((size_t)TOT * HEADS * 4);
  bf16_t* wb     = (bf16_t*)carve((size_t)H * H * 2);
  bf16_t* xb     = (bf16_t*)carve((size_t)N * H * 2);
  float*  tA     = (float*)carve((size_t)N * H * 4);
  float*  tB     = (float*)carve((size_t)N * H * 4);
  float*  acc    = (float*)carve((size_t)N * H * 4);

  const unsigned TB = 256;
  const unsigned gNH   = cdiv((long long)N * H, TB);
  const unsigned gEdgeW = cdiv((long long)TOT * 32, TB);   // one wave per edge

  // ---- degree -> dinv (self loops: init deg = 1)
  fill_f32_kernel<<<cdiv(N, TB), TB, 0, stream>>>(dinv_a, 1.0f, N);
  fill_f32_kernel<<<cdiv(N, TB), TB, 0, stream>>>(dinv_r, 1.0f, N);
  deg_accum_kernel<<<cdiv(E, TB), TB, 0, stream>>>(dst_a, dinv_a, E);
  deg_accum_kernel<<<cdiv(E, TB), TB, 0, stream>>>(dst_r, dinv_r, E);
  finalize_dinv_kernel<<<cdiv(N, TB), TB, 0, stream>>>(dinv_a, N);
  finalize_dinv_kernel<<<cdiv(N, TB), TB, 0, stream>>>(dinv_r, N);

  // ---- layer 1: two GCNConvs + BN + SiLU  (b1/bo biases cancel inside BN)
  f32_to_bf16_kernel<<<gNH, TB, 0, stream>>>(x, xb, N * H);
  const unsigned gGemmH = cdiv((long long)(N / 32) * (H / 32), 8);
  convert_w_transpose_kernel<<<cdiv((long long)H * H, TB), TB, 0, stream>>>(W1a, wb, H, H);
  gemm_bf16_wmma_kernel<<<gGemmH, TB, 0, stream>>>(xb, wb, nullptr, tA, N, H, H);
  convert_w_transpose_kernel<<<cdiv((long long)H * H, TB), TB, 0, stream>>>(W1r, wb, H, H);
  gemm_bf16_wmma_kernel<<<gGemmH, TB, 0, stream>>>(xb, wb, nullptr, tB, N, H, H);
  fill_f32_kernel<<<gNH, TB, 0, stream>>>(acc, 0.f, N * H);
  gcn_scatter_kernel<<<gEdgeW, TB, 0, stream>>>(tA, dinv_a, src_a, dst_a, acc, E, N, H);
  gcn_scatter_kernel<<<gEdgeW, TB, 0, stream>>>(tB, dinv_r, src_r, dst_r, acc, E, N, H);
  fill_f32_kernel<<<cdiv(2 * H, TB), TB, 0, stream>>>(stats, 0.f, 2 * H);
  bn_reduce_kernel<<<cdiv(N, 50), 512, 0, stream>>>(acc, stats, N);
  bn_silu_bf16_kernel<<<gNH, TB, 0, stream>>>(acc, stats, g1, be1, xb, N);   // h1 (bf16) -> xb

  // ---- layer 2: two GATv2Convs, processed sequentially into `acc`
  fill_f32_kernel<<<gNH, TB, 0, stream>>>(acc, 0.f, N * H);
  const unsigned gEH = cdiv((long long)TOT * HEADS, TB);
  for (int t = 0; t < 2; ++t) {
    const float *Wl = t ? Wlr : Wla, *bl = t ? blr : bla;
    const float *Wr = t ? Wrr : Wra, *br = t ? brr : bra;
    const float *at = t ? attr : atta;
    const int *sp = t ? src_r : src_a, *dp = t ? dst_r : dst_a;
    const float *dv = t ? dinv_r : dinv_a; (void)dv;
    convert_w_transpose_kernel<<<cdiv((long long)H * H, TB), TB, 0, stream>>>(Wl, wb, H, H);
    gemm_bf16_wmma_kernel<<<gGemmH, TB, 0, stream>>>(xb, wb, bl, tA, N, H, H);   // hl
    convert_w_transpose_kernel<<<cdiv((long long)H * H, TB), TB, 0, stream>>>(Wr, wb, H, H);
    gemm_bf16_wmma_kernel<<<gGemmH, TB, 0, stream>>>(xb, wb, br, tB, N, H, H);   // hr
    fill_f32_kernel<<<cdiv(N * HEADS, TB), TB, 0, stream>>>((float*)mEnc, 0.f, N * HEADS); // enc(-inf)>=0
    fill_f32_kernel<<<cdiv(N * HEADS, TB), TB, 0, stream>>>(denom, 0.f, N * HEADS);
    gat_logits_kernel<<<gEdgeW, TB, 0, stream>>>(tA, tB, at, sp, dp, E, N, logits, mEnc);
    gat_exp_denom_kernel<<<gEH, TB, 0, stream>>>(logits, mEnc, denom, dp, E, N);
    gat_scatter_kernel<<<gEdgeW, TB, 0, stream>>>(tA, logits, denom, sp, dp, E, N, acc);
  }
  fill_f32_kernel<<<cdiv(2 * H, TB), TB, 0, stream>>>(stats, 0.f, 2 * H);
  bn_reduce_kernel<<<cdiv(N, 50), 512, 0, stream>>>(acc, stats, N);
  bn_silu_bf16_kernel<<<gNH, TB, 0, stream>>>(acc, stats, g2, be2, xb, N);   // h2 (bf16) -> xb

  // ---- latent projection: two GCNConvs into d_out (+ bz biases)
  float* z = (float*)d_out;
  const unsigned gGemmL = cdiv((long long)(N / 32) * (L / 32), 8);
  convert_w_transpose_kernel<<<cdiv((long long)H * L, TB), TB, 0, stream>>>(Wza, wb, H, L);
  gemm_bf16_wmma_kernel<<<gGemmL, TB, 0, stream>>>(xb, wb, nullptr, tA, N, H, L);
  convert_w_transpose_kernel<<<cdiv((long long)H * L, TB), TB, 0, stream>>>(Wzr, wb, H, L);
  gemm_bf16_wmma_kernel<<<gGemmL, TB, 0, stream>>>(xb, wb, nullptr, tB, N, H, L);
  init_out_bias_kernel<<<cdiv((long long)N * L, TB), TB, 0, stream>>>(z, bza, bzr, N * L);
  gcn_scatter_kernel<<<gEdgeW, TB, 0, stream>>>(tA, dinv_a, src_a, dst_a, z, E, N, L);
  gcn_scatter_kernel<<<gEdgeW, TB, 0, stream>>>(tB, dinv_r, src_r, dst_r, z, E, N, L);
  (void)n_in; (void)out_size; (void)ws_size;
}
